// RowAttentionWithPairBias_73254962201330
// MI455X (gfx1250) — matrix operand order repaired and checked by reading, not compiled, not run on hardware
//
#include <hip/hip_runtime.h>
#include <hip/hip_bf16.h>
#include <math.h>

typedef __attribute__((ext_vector_type(16))) _Float16 v16h;
typedef __attribute__((ext_vector_type(8)))  _Float16 v8h;
typedef __attribute__((ext_vector_type(8)))  float    v8f;

#define ROWS   32768   // B*M*N
#define INC    256     // IN_C
#define CC     32      // ATTN_C
#define HH     8       // HEADS
#define MM     128
#define NN     256
#define PAIRC  128
#define QKV_COLS 768
#define TOT_COLS 1024

static __device__ __forceinline__ v8f wmma_f16(v16h a, v16h b, v8f c) {
  // (neg_a, A, neg_b, B, c_mod, C, reuse_a, reuse_b)
  return __builtin_amdgcn_wmma_f32_16x16x32_f16(false, a, false, b, (short)0, c,
                                                false, false);
}

static __device__ __forceinline__ v16h pack16(v8h lo, v8h hi) {
  return __builtin_shufflevector(lo, hi, 0, 1, 2, 3, 4, 5, 6, 7,
                                 8, 9, 10, 11, 12, 13, 14, 15);
}

// ---------------------------------------------------------------------------
// K0: one-shot fp32 -> f16 weight conversion.
//     wcomb[1024][256] = concat(qkv_w, gate_w); fw16[256][256] = final_w
// ---------------------------------------------------------------------------
__global__ __launch_bounds__(256) void convert_weights(
    const float* __restrict__ qkv_w, const float* __restrict__ gate_w,
    const float* __restrict__ final_w, _Float16* __restrict__ wcomb,
    _Float16* __restrict__ fw16) {
  int idx = blockIdx.x * 256 + threadIdx.x;
  const int NCOMB = TOT_COLS * INC;          // 262144
  if (idx < NCOMB) {
    int col = idx >> 8;
    int k = idx & 255;
    float v = (col < QKV_COLS) ? qkv_w[(size_t)col * INC + k]
                               : gate_w[(size_t)(col - QKV_COLS) * INC + k];
    wcomb[idx] = (_Float16)v;
  } else {
    int fidx = idx - NCOMB;                  // [0, 65536)
    fw16[fidx] = (_Float16)final_w[fidx];
  }
}

// ---------------------------------------------------------------------------
// K1: LayerNorm over IN_C=256, one wave per row, output f16
// ---------------------------------------------------------------------------
__global__ __launch_bounds__(256) void ln_kernel(
    const float* __restrict__ x, const float* __restrict__ w,
    const float* __restrict__ b, _Float16* __restrict__ xn) {
  int row = (blockIdx.x * blockDim.x + threadIdx.x) >> 5;
  int lane = threadIdx.x & 31;
  if (row >= ROWS) return;
  const float4* r4 = (const float4*)(x + (size_t)row * INC);
  float4 p0 = r4[lane * 2];
  float4 p1 = r4[lane * 2 + 1];
  float v[8] = {p0.x, p0.y, p0.z, p0.w, p1.x, p1.y, p1.z, p1.w};
  float s = 0.f;
  for (int i = 0; i < 8; ++i) s += v[i];
  for (int m = 16; m >= 1; m >>= 1) s += __shfl_xor(s, m, 32);
  float mu = s * (1.0f / 256.0f);
  float vs = 0.f;
  for (int i = 0; i < 8; ++i) { v[i] -= mu; vs += v[i] * v[i]; }
  for (int m = 16; m >= 1; m >>= 1) vs += __shfl_xor(vs, m, 32);
  float rinv = rsqrtf(vs * (1.0f / 256.0f) + 1e-5f);
  _Float16* orow = xn + (size_t)row * INC;
  for (int i = 0; i < 8; ++i) {
    int c = lane * 8 + i;
    orow[c] = (_Float16)(v[i] * rinv * w[c] + b[c]);
  }
}

// ---------------------------------------------------------------------------
// K2: fused QKV + gate GEMM:  xn(32768x256) @ wcomb^T(256x1024), WMMA f16.
//     Each wave: 1 row-tile x 4 col-tiles (A fragments reused in registers).
//     Epilogue scatters q/k/v into [m][h][n][c] f16 and sigmoid-gate to gateb.
// ---------------------------------------------------------------------------
__global__ __launch_bounds__(256) void qkv_gate_gemm(
    const _Float16* __restrict__ xn, const _Float16* __restrict__ wcomb,
    const float* __restrict__ gate_b, _Float16* __restrict__ qb,
    _Float16* __restrict__ kb, _Float16* __restrict__ vb,
    _Float16* __restrict__ gateb) {
  int job = blockIdx.x * 8 + (threadIdx.x >> 5);
  if (job >= 2048 * 16) return;
  int cgroup = job & 15;           // 4 col-tiles per group
  int rtile = job >> 4;
  int lane = threadIdx.x & 31;
  int l16 = lane & 15, hi = lane >> 4;

  // A fragments: 8 k-chunks, kept live across the 4 column tiles
  const _Float16* arow = xn + ((size_t)rtile * 16 + l16) * INC;
  v16h a[8];
  for (int kc = 0; kc < 8; ++kc) {
    v8h alo = *(const v8h*)(arow + kc * 32 + hi * 16);
    v8h ahi = *(const v8h*)(arow + kc * 32 + hi * 16 + 8);
    a[kc] = pack16(alo, ahi);
  }

  for (int cg = 0; cg < 4; ++cg) {
    int col = (cgroup * 4 + cg) * 16 + l16;             // B column = l16
    const _Float16* wrow = wcomb + (size_t)col * INC;
    v8f acc = {};
    for (int kc = 0; kc < 8; ++kc) {
      v16h bfr = *(const v16h*)(wrow + kc * 32 + hi * 16);  // 32B direct load
      acc = wmma_f16(a[kc], bfr, acc);
    }
    for (int r = 0; r < 8; ++r) {
      int row = rtile * 16 + r + hi * 8;  // D element (M=r+8*hi, N=l16)
      int m = row >> 8;
      int n = row & 255;
      float v = acc[r];
      if (col < QKV_COLS) {
        int c = col / 24;        // feature -> (c, t) with 3H=24
        int t = col - c * 24;
        int h = t & 7;
        int kind = t >> 3;       // 0=q 1=k 2=v
        _Float16* dst = (kind == 0) ? qb : (kind == 1 ? kb : vb);
        dst[(((size_t)(m * HH + h)) * NN + n) * CC + c] = (_Float16)v;
      } else {
        int f = col - QKV_COLS;
        float g = 1.0f / (1.0f + expf(-(v + gate_b[f])));
        gateb[(size_t)row * 256 + f] = (_Float16)g;
      }
    }
  }
}

// ---------------------------------------------------------------------------
// K3: pair bias  bias[h][i][j] = (1/sqrt(32)) * dot(x2d[i][j][:], x2d_w[h][:])
//     one wave per (i,j) pair
// ---------------------------------------------------------------------------
__global__ __launch_bounds__(256) void pair_bias(
    const float* __restrict__ x2d, const float* __restrict__ x2d_w,
    float* __restrict__ biasb) {
  int pair = blockIdx.x * 8 + (threadIdx.x >> 5);
  if (pair >= NN * NN) return;
  int lane = threadIdx.x & 31;
  const float4 xv = ((const float4*)(x2d + (size_t)pair * PAIRC))[lane];
  float part[8];
  for (int h = 0; h < 8; ++h) {
    const float4 wv = ((const float4*)(x2d_w + (size_t)h * PAIRC))[lane];
    float p = xv.x * wv.x + xv.y * wv.y + xv.z * wv.z + xv.w * wv.w;
    for (int msk = 16; msk >= 1; msk >>= 1) p += __shfl_xor(p, msk, 32);
    part[h] = p;
  }
  if (lane == 0) {
    const float factor = 0.17677669529663687f;  // 1/sqrt(32)
    for (int h = 0; h < 8; ++h)
      biasb[(size_t)h * (NN * NN) + pair] = part[h] * factor;
  }
}

// ---------------------------------------------------------------------------
// K4: attention per (m, h, i-half). 8 waves, each owns a 16-row query tile.
//     S = Q K^T (one WMMA per 16x16 tile, K=32), +bias, f32 softmax over j,
//     O = P V via WMMA with V^T staged in LDS. Gated f16 output to attnb.
// ---------------------------------------------------------------------------
__global__ __launch_bounds__(256) void attention(
    const _Float16* __restrict__ qb, const _Float16* __restrict__ kb,
    const _Float16* __restrict__ vb, const float* __restrict__ biasb,
    const _Float16* __restrict__ gateb, _Float16* __restrict__ attnb) {
  __shared__ _Float16 Vt[CC][NN];       // 16 KB, V transposed
  __shared__ _Float16 Pst[8][16][32];   // 8 KB, per-wave P restage
  int bid = blockIdx.x;
  int ihalf = bid & 1;
  int h = (bid >> 1) & 7;
  int m = bid >> 4;
  int wv = threadIdx.x >> 5;
  int lane = threadIdx.x & 31;
  int l16 = lane & 15, hi = lane >> 4;
  size_t mhbase = ((size_t)(m * HH + h)) * NN;

  // stage V^T into LDS: each thread does 4 x b128 loads (8 halfs = 8 c-values
  // of one n-row), transposed scatter on the DS-store side only.
  const _Float16* vbase = vb + mhbase * CC;
  for (int t = 0; t < 4; ++t) {
    int chunk = threadIdx.x + t * 256;       // [0,1024): n = chunk>>2, c0 = (chunk&3)*8
    int n = chunk >> 2;
    int c0 = (chunk & 3) * 8;
    v8h vv = *(const v8h*)(vbase + (size_t)n * CC + c0);
    for (int i = 0; i < 8; ++i) Vt[c0 + i][n] = vv[i];
  }
  __syncthreads();

  // Q A-fragment (rows i0..i0+15), loaded directly in register layout
  int i0 = (ihalf * 8 + wv) * 16;
  const _Float16* qrow = qb + (mhbase + i0 + l16) * CC;
  v16h aQ = pack16(*(const v8h*)(qrow + hi * 16),
                   *(const v8h*)(qrow + hi * 16 + 8));

  // S = Q K^T + bias : 16 j-tiles, one wmma each (K dim == 32)
  v8f S[16];
  const float* biash = biasb + (size_t)h * NN * NN;
  for (int jt = 0; jt < 16; ++jt) {
    const _Float16* krow = kb + (mhbase + (size_t)jt * 16 + l16) * CC;
    v16h bK = *(const v16h*)(krow + hi * 16);  // B col = K row j, 32B load
    v8f s = {};
    s = wmma_f16(aQ, bK, s);
    for (int r = 0; r < 8; ++r) {
      int i = i0 + r + hi * 8;
      int j = jt * 16 + l16;
      s[r] += biash[(size_t)i * NN + j];
    }
    S[jt] = s;
  }

  // softmax over j (row spread across 16 lanes of same half-wave + 16 tiles)
  float rmax[8], rsum[8], rinv[8];
  for (int r = 0; r < 8; ++r) rmax[r] = -3.0e38f;
  for (int jt = 0; jt < 16; ++jt)
    for (int r = 0; r < 8; ++r) rmax[r] = fmaxf(rmax[r], S[jt][r]);
  for (int r = 0; r < 8; ++r)
    for (int msk = 8; msk >= 1; msk >>= 1)
      rmax[r] = fmaxf(rmax[r], __shfl_xor(rmax[r], msk, 32));
  for (int r = 0; r < 8; ++r) rsum[r] = 0.f;
  for (int jt = 0; jt < 16; ++jt)
    for (int r = 0; r < 8; ++r) {
      float e = expf(S[jt][r] - rmax[r]);
      S[jt][r] = e;
      rsum[r] += e;
    }
  for (int r = 0; r < 8; ++r) {
    for (int msk = 8; msk >= 1; msk >>= 1)
      rsum[r] += __shfl_xor(rsum[r], msk, 32);
    rinv[r] = 1.0f / rsum[r];
  }

  // O = P @ V : 8 j-chunks of 32; restage P (C-layout -> A-layout) via LDS.
  // Each wave touches only Pst[wv]; LDS pipe is in-order within a wave.
  v8f O[2] = {};
  for (int jc = 0; jc < 8; ++jc) {
    for (int t = 0; t < 2; ++t) {
      int jt = jc * 2 + t;
      for (int r = 0; r < 8; ++r)
        Pst[wv][r + hi * 8][t * 16 + l16] = (_Float16)(S[jt][r] * rinv[r]);
    }
    const _Float16* prow = &Pst[wv][l16][0];
    v16h aP = pack16(*(const v8h*)(prow + hi * 16),
                     *(const v8h*)(prow + hi * 16 + 8));
    for (int ct = 0; ct < 2; ++ct) {
      v16h bV = *(const v16h*)(&Vt[ct * 16 + l16][jc * 32 + hi * 16]);
      O[ct] = wmma_f16(aP, bV, O[ct]);
    }
  }

  // gate and store f16 in [row][c*H + h]
  size_t rowbase = (size_t)m * NN;
  for (int ct = 0; ct < 2; ++ct)
    for (int r = 0; r < 8; ++r) {
      int i = i0 + r + hi * 8;
      int c = ct * 16 + l16;
      size_t row = rowbase + i;
      int f = c * HH + h;
      float g = (float)gateb[row * 256 + f];
      attnb[row * 256 + f] = (_Float16)(O[ct][r] * g);
    }
}

// ---------------------------------------------------------------------------
// K5: final projection  attnb(32768x256) @ fw16^T(256x256) + final_b
//     Each wave: 1 row-tile x 4 col-tiles (A fragments reused).
// ---------------------------------------------------------------------------
__global__ __launch_bounds__(256) void final_gemm(
    const _Float16* __restrict__ attnb, const _Float16* __restrict__ fw16,
    const float* __restrict__ fb, float* __restrict__ out) {
  int job = blockIdx.x * 8 + (threadIdx.x >> 5);
  if (job >= 2048 * 4) return;
  int cgroup = job & 3;            // 4 col-tiles per group, 16 total
  int rtile = job >> 2;
  int lane = threadIdx.x & 31;
  int l16 = lane & 15, hi = lane >> 4;

  const _Float16* arow = attnb + ((size_t)rtile * 16 + l16) * 256;
  v16h a[8];
  for (int kc = 0; kc < 8; ++kc) {
    v8h alo = *(const v8h*)(arow + kc * 32 + hi * 16);
    v8h ahi = *(const v8h*)(arow + kc * 32 + hi * 16 + 8);
    a[kc] = pack16(alo, ahi);
  }

  for (int cg = 0; cg < 4; ++cg) {
    int o = (cgroup * 4 + cg) * 16 + l16;
    const _Float16* wrow = fw16 + (size_t)o * 256;
    v8f acc = {};
    for (int kc = 0; kc < 8; ++kc) {
      v16h bfr = *(const v16h*)(wrow + kc * 32 + hi * 16);
      acc = wmma_f16(a[kc], bfr, acc);
    }
    float bias = fb[o];
    for (int r = 0; r < 8; ++r) {
      int row = rtile * 16 + r + hi * 8;
      out[(size_t)row * 256 + o] = acc[r] + bias;
    }
  }
}

// ---------------------------------------------------------------------------
extern "C" void kernel_launch(void* const* d_in, const int* in_sizes, int n_in,
                              void* d_out, int out_size, void* d_ws,
                              size_t ws_size, hipStream_t stream) {
  (void)in_sizes; (void)n_in; (void)out_size; (void)ws_size;
  const float* x1d    = (const float*)d_in[0];
  const float* x2d    = (const float*)d_in[1];
  const float* norm_w = (const float*)d_in[2];
  const float* norm_b = (const float*)d_in[3];
  const float* qkv_w  = (const float*)d_in[4];
  const float* x2d_w  = (const float*)d_in[5];
  const float* gate_w = (const float*)d_in[6];
  const float* gate_b = (const float*)d_in[7];
  const float* final_w= (const float*)d_in[8];
  const float* final_b= (const float*)d_in[9];
  float* out = (float*)d_out;

  char* ws = (char*)d_ws;
  const size_t SZ = (size_t)ROWS * 256 * sizeof(_Float16);  // 16 MB each
  _Float16* xn    = (_Float16*)(ws);
  _Float16* qb    = (_Float16*)(ws + SZ);
  _Float16* kb    = (_Float16*)(ws + 2 * SZ);
  _Float16* vb    = (_Float16*)(ws + 3 * SZ);
  _Float16* gateb = (_Float16*)(ws + 4 * SZ);
  float*    biasb = (float*)(ws + 5 * SZ);
  size_t off = 5 * SZ + (size_t)HH * NN * NN * sizeof(float);
  _Float16* attnb = (_Float16*)(ws + off);
  off += SZ;
  _Float16* wcomb = (_Float16*)(ws + off);                 // 1024*256 f16
  off += (size_t)TOT_COLS * INC * sizeof(_Float16);
  _Float16* fw16  = (_Float16*)(ws + off);                 // 256*256 f16

  convert_weights<<<(TOT_COLS * INC + 256 * 256) / 256, 256, 0, stream>>>(
      qkv_w, gate_w, final_w, wcomb, fw16);
  ln_kernel<<<ROWS / 8, 256, 0, stream>>>(x1d, norm_w, norm_b, xn);
  qkv_gate_gemm<<<(2048 * 16) / 8, 256, 0, stream>>>(xn, wcomb, gate_b, qb, kb,
                                                     vb, gateb);
  pair_bias<<<(NN * NN) / 8, 256, 0, stream>>>(x2d, x2d_w, biasb);
  attention<<<MM * HH * 2, 256, 0, stream>>>(qb, kb, vb, biasb, gateb, attnb);
  final_gemm<<<(2048 * 4) / 8, 256, 0, stream>>>(attnb, fw16, final_b, out);
}
